// MultiHeadAttention_39084202393720
// MI455X (gfx1250) — matrix operand order repaired and checked by reading, not compiled, not run on hardware
//
#include <hip/hip_runtime.h>

typedef __attribute__((ext_vector_type(16))) __bf16 v16bf;
typedef __attribute__((ext_vector_type(8)))  __bf16 v8bf;
typedef __attribute__((ext_vector_type(4)))  __bf16 v4bf;
typedef __attribute__((ext_vector_type(8)))  float  v8f;
typedef __attribute__((ext_vector_type(4)))  unsigned int v4ui;
typedef __attribute__((ext_vector_type(8)))  int v8si;
typedef __attribute__((ext_vector_type(4)))  int v4si;

#define D_MODEL 1024
#define S_LEN   2048
#define NHEAD   16
#define HDIM    64
#define BATCH   4
#define MROWS   (BATCH * S_LEN)   // 8192

union FragU { v16bf v; v8bf h[2]; };

// ---------------------------------------------------------------------------
// A-fragment (16x32, bf16, row-major source). Contiguous k order (matches load_b_frag):
// lane L: row = L%16, g = L/16; halfs h -> K = 16g + h
// ---------------------------------------------------------------------------
__device__ __forceinline__ v16bf load_a_frag(const __bf16* __restrict__ base,
                                             int ld, int m0, int k0, int lane) {
  const int r = lane & 15, g = lane >> 4;
  const __bf16* p = base + (size_t)(m0 + r) * ld + k0 + g * 16;
  FragU u;
  u.h[0] = *(const v8bf*)p;
  u.h[1] = *(const v8bf*)(p + 8);
  return u.v;
}

// ---------------------------------------------------------------------------
// B-fragment (32x16, bf16) from an [N, K] row-major source (W^T / K^T / V^T).
// lane L: col = L%16, g = L/16; halfs h -> K = 16g + h  => one 32B load.
// ---------------------------------------------------------------------------
__device__ __forceinline__ v16bf load_b_frag(const __bf16* __restrict__ base,
                                             int ld, int n0, int k0, int lane) {
  const int c = lane & 15, g = lane >> 4;
  return *(const v16bf*)(base + (size_t)(n0 + c) * ld + k0 + g * 16);
}

__device__ __forceinline__ v8f wmma_bf16(v16bf a, v16bf b, v8f c) {
  return __builtin_amdgcn_wmma_f32_16x16x32_bf16(false, a, false, b,
                                                 (short)0, c, false, false);
}

__device__ __forceinline__ v8f vzero8() {
  v8f z = {0.f, 0.f, 0.f, 0.f, 0.f, 0.f, 0.f, 0.f};
  return z;
}

// ---------------------------------------------------------------------------
// TDM: 2D tile (bf16) global -> LDS. Descriptor per CDNA5 ISA ch.8:
//  group0: count=1 | lds_addr | global_addr[56:0] | type=2
//  group1: data_size=1 (2B), tensor dims, tile dims, dim0 stride (elem units)
// Groups 2/3 (and trailing group arg) zero: 2D tensor, no iterate/gather.
// ---------------------------------------------------------------------------
__device__ __forceinline__ void tdm_load_2d_bf16(
    unsigned int lds_off, const __bf16* gptr,
    unsigned int tensor_d0, unsigned int tensor_d1, unsigned long long stride0,
    unsigned int tile_d0, unsigned int tile_d1) {
  unsigned long long ga = (unsigned long long)(size_t)gptr;
  v4ui g0;
  g0[0] = 1u;                                         // count=1, user mode
  g0[1] = lds_off;                                    // LDS byte address
  g0[2] = (unsigned int)ga;                           // global_addr[31:0]
  g0[3] = (unsigned int)((ga >> 32) & 0x1FFFFFFull)   // global_addr[56:32]
          | (2u << 30);                               // type = 2 ("image")
  v8si g1;
  g1[0] = (int)(1u << 16);                            // data_size=1 -> 2 bytes
  g1[1] = (int)((tensor_d0 & 0xFFFFu) << 16);         // tensor_dim0[15:0]
  g1[2] = (int)((tensor_d0 >> 16) | ((tensor_d1 & 0xFFFFu) << 16));
  g1[3] = (int)((tensor_d1 >> 16) | (tile_d0 << 16)); // tile_dim0
  g1[4] = (int)(tile_d1 & 0xFFFFu);                   // tile_dim1 (tile_dim2=0)
  g1[5] = (int)(unsigned int)stride0;                 // dim0_stride[31:0]
  g1[6] = (int)((stride0 >> 32) & 0xFFFFull);         // dim0_stride[47:32]
  g1[7] = 0;
  v4si gz4 = {0, 0, 0, 0};
  v8si gz8 = {0, 0, 0, 0, 0, 0, 0, 0};
  __builtin_amdgcn_tensor_load_to_lds(g0, g1, gz4, gz4, gz8, 0);
}

// ---------------------------------------------------------------------------
// fp32 -> bf16 bulk convert (4 elems/thread)
// ---------------------------------------------------------------------------
__global__ __launch_bounds__(256) void cvt_kernel(const float* __restrict__ s,
                                                  __bf16* __restrict__ d, int n4) {
  int i = blockIdx.x * blockDim.x + threadIdx.x;
  if (i >= n4) return;
  float4 f = ((const float4*)s)[i];
  v4bf o = { (__bf16)f.x, (__bf16)f.y, (__bf16)f.z, (__bf16)f.w };
  ((v4bf*)d)[i] = o;
}

// ---------------------------------------------------------------------------
// Fused QKV projection. Each wave: 16(M) x 64(N) tile -> A-frag reused 4x.
// Q,K stored [B,H,S,64]; V stored transposed [B,H,64,S].
// blockIdx.y covers one head (64 output cols).
// ---------------------------------------------------------------------------
__global__ __launch_bounds__(128) void qkv_kernel(
    const __bf16* __restrict__ xb,
    const __bf16* __restrict__ wqb, const __bf16* __restrict__ wkb,
    const __bf16* __restrict__ wvb,
    const float* __restrict__ bq, const float* __restrict__ bk,
    const float* __restrict__ bv,
    __bf16* __restrict__ Qb, __bf16* __restrict__ Kb, __bf16* __restrict__ Vtb) {
  const int lane = threadIdx.x & 31;
  const int wave = threadIdx.x >> 5;
  const int m0 = (blockIdx.x * 4 + wave) * 16;
  const int n0 = blockIdx.y * 64;                  // one head
  const int z  = blockIdx.z;
  const __bf16* W   = (z == 0) ? wqb : (z == 1) ? wkb : wvb;
  const float* bias = (z == 0) ? bq  : (z == 1) ? bk  : bv;

  v8f acc[4];
#pragma unroll
  for (int s = 0; s < 4; ++s) acc[s] = vzero8();

#pragma unroll 2
  for (int k0 = 0; k0 < D_MODEL; k0 += 32) {
    v16bf a = load_a_frag(xb, D_MODEL, m0, k0, lane);
#pragma unroll
    for (int s = 0; s < 4; ++s) {
      v16bf b = load_b_frag(W, D_MODEL, n0 + s * 16, k0, lane);
      acc[s] = wmma_bf16(a, b, acc[s]);
    }
  }

  const int c = lane & 15, g = lane >> 4;
  const int h = blockIdx.y;
#pragma unroll
  for (int s = 0; s < 4; ++s) {
    const int d = s * 16 + c;                      // dim within head
    const float bn = bias[n0 + d];
#pragma unroll
    for (int vr = 0; vr < 8; ++vr) {
      const int m = m0 + vr + 8 * g;
      const int bi = m >> 11, srow = m & (S_LEN - 1);
      const __bf16 val = (__bf16)(acc[s][vr] + bn);
      const size_t hb = (size_t)bi * NHEAD + h;
      if (z == 0)      Qb [(hb * S_LEN + srow) * HDIM + d] = val;
      else if (z == 1) Kb [(hb * S_LEN + srow) * HDIM + d] = val;
      else             Vtb[(hb * HDIM + d) * S_LEN + srow] = val;
    }
  }
}

// ---------------------------------------------------------------------------
// Flash attention (causal). Block = 8 waves = 128 query rows of one (b,h).
// Double-buffered TDM pipeline: wave 0 issues the NEXT 32-key K/V tiles into
// the alternate LDS buffers, then waits TENSORcnt<=2 so the CURRENT tile's
// two (in-order) TDM loads are complete; barrier; all waves run QK^T (4 WMMA),
// online softmax, and PV (4 WMMA) off LDS.
// ---------------------------------------------------------------------------
__global__ __launch_bounds__(256) void attn_kernel(
    const __bf16* __restrict__ Qb, const __bf16* __restrict__ Kb,
    const __bf16* __restrict__ Vtb, __bf16* __restrict__ ctxb) {
  __shared__ __bf16 Kt[2][32 * HDIM];    // [buf][key][d]   2 x 4KB
  __shared__ __bf16 Vs[2][HDIM * 32];    // [buf][d][key]   2 x 4KB
  __shared__ __bf16 Plds[8][16 * 32];    // per-wave P      8KB

  const int lane = threadIdx.x & 31;
  const int wave = threadIdx.x >> 5;
  const int bh = blockIdx.y;                       // b*16 + h
  const int q0blk = blockIdx.x * 128;
  const int q0 = q0blk + wave * 16;
  const int c = lane & 15, g = lane >> 4;

  const __bf16* Q  = Qb  + (size_t)bh * S_LEN * HDIM;
  const __bf16* K  = Kb  + (size_t)bh * S_LEN * HDIM;
  const __bf16* Vt = Vtb + (size_t)bh * HDIM * S_LEN;

  const unsigned int kt0 = (unsigned int)(size_t)(void*)&Kt[0][0];
  const unsigned int vs0 = (unsigned int)(size_t)(void*)&Vs[0][0];
  const unsigned int BUFB = 32 * HDIM * 2;         // bytes per buffer

  // Q fragments for both 32-wide chunks of Hd=64, kept in registers
  const v16bf qa0 = load_a_frag(Q, HDIM, q0, 0,  lane);
  const v16bf qa1 = load_a_frag(Q, HDIM, q0, 32, lane);

  float m_i[8], l_i[8];
  v8f acc[4];
#pragma unroll
  for (int r = 0; r < 8; ++r) { m_i[r] = -__builtin_inff(); l_i[r] = 0.f; }
#pragma unroll
  for (int d = 0; d < 4; ++d) acc[d] = vzero8();

  const float scale = 0.125f;                      // 1/sqrt(64)
  const int nt = (q0blk + 128) / 32;               // causal tile count (block)

  if (wave == 0) {                                 // prologue: tile 0 -> buf 0
    tdm_load_2d_bf16(kt0, K, HDIM, S_LEN, HDIM, HDIM, 32);
    tdm_load_2d_bf16(vs0, Vt, S_LEN, HDIM, S_LEN, 32, HDIM);
  }

  for (int i = 0; i < nt; ++i) {
    const int t0 = i * 32;
    const int buf = i & 1;
    if (wave == 0) {
      if (i + 1 < nt) {                            // prefetch tile i+1
        const int t1 = t0 + 32;
        const unsigned int nb = (unsigned int)(buf ^ 1) * BUFB;
        tdm_load_2d_bf16(kt0 + nb, K + (size_t)t1 * HDIM,
                         HDIM, S_LEN, HDIM, HDIM, 32);
        tdm_load_2d_bf16(vs0 + nb, Vt + t1,
                         S_LEN, HDIM, S_LEN, 32, HDIM);
        __builtin_amdgcn_s_wait_tensorcnt(2);      // tile i complete
      } else {
        __builtin_amdgcn_s_wait_tensorcnt(0);      // last tile complete
      }
    }
    __syncthreads();
    const __bf16* Ktb = &Kt[buf][0];
    const __bf16* Vsb = &Vs[buf][0];

    // ---- scores: two 16x16 subtiles, K-frags from LDS ----
    v8f s0 = vzero8(), s1 = vzero8();
    {
      v16bf kb0 = *(const v16bf*)(&Ktb[(c) * HDIM + 0  + g * 16]);
      v16bf kb1 = *(const v16bf*)(&Ktb[(c) * HDIM + 32 + g * 16]);
      s0 = wmma_bf16(qa0, kb0, s0);
      s0 = wmma_bf16(qa1, kb1, s0);
      v16bf kb2 = *(const v16bf*)(&Ktb[(16 + c) * HDIM + 0  + g * 16]);
      v16bf kb3 = *(const v16bf*)(&Ktb[(16 + c) * HDIM + 32 + g * 16]);
      s1 = wmma_bf16(qa0, kb2, s1);
      s1 = wmma_bf16(qa1, kb3, s1);
    }

    // ---- online softmax in C-layout ----
    __bf16* P = Plds[wave];
#pragma unroll
    for (int vr = 0; vr < 8; ++vr) {
      const int qrow = q0 + vr + 8 * g;
      float v0 = s0[vr] * scale;
      float v1 = s1[vr] * scale;
      if (t0 + c > qrow)      v0 = -__builtin_inff();
      if (t0 + 16 + c > qrow) v1 = -__builtin_inff();
      float mx = fmaxf(m_i[vr], fmaxf(v0, v1));
#pragma unroll
      for (int off = 1; off < 16; off <<= 1)
        mx = fmaxf(mx, __shfl_xor(mx, off, 32));
      const float rescale = __expf(m_i[vr] - mx);
      const float p0 = __expf(v0 - mx);
      const float p1 = __expf(v1 - mx);
      float rs = p0 + p1;
#pragma unroll
      for (int off = 1; off < 16; off <<= 1)
        rs += __shfl_xor(rs, off, 32);
      l_i[vr] = l_i[vr] * rescale + rs;
      m_i[vr] = mx;
      acc[0][vr] *= rescale; acc[1][vr] *= rescale;
      acc[2][vr] *= rescale; acc[3][vr] *= rescale;
      P[(vr + 8 * g) * 32 + c]      = (__bf16)p0;
      P[(vr + 8 * g) * 32 + 16 + c] = (__bf16)p1;
    }

    // ---- P (C-layout in LDS) re-read as A-fragment; DS in-order per wave ----
    FragU pu;
    const __bf16* pr = P + c * 32 + g * 16;
    pu.h[0] = *(const v8bf*)pr;
    pu.h[1] = *(const v8bf*)(pr + 8);

    // ---- ctx += P @ V, V-frags from LDS [d][key] ----
#pragma unroll
    for (int d = 0; d < 4; ++d) {
      v16bf vb = *(const v16bf*)(&Vsb[(d * 16 + c) * 32 + g * 16]);
      acc[d] = wmma_bf16(pu.v, vb, acc[d]);
    }
    __syncthreads();   // all waves done with buf before TDM re-targets it
  }

  // epilogue: normalize, store ctx as [B, S, D] bf16
  const int b = bh >> 4, h = bh & 15;
#pragma unroll
  for (int vr = 0; vr < 8; ++vr) {
    const float inv = 1.f / l_i[vr];
    const int srow = q0 + vr + 8 * g;
    const size_t rowbase = ((size_t)b * S_LEN + srow) * D_MODEL + h * HDIM;
#pragma unroll
    for (int d = 0; d < 4; ++d)
      ctxb[rowbase + d * 16 + c] = (__bf16)(acc[d][vr] * inv);
  }
}

// ---------------------------------------------------------------------------
// Output projection: 16x64 tile per wave, fp32 out.
// ---------------------------------------------------------------------------
__global__ __launch_bounds__(128) void oproj_kernel(
    const __bf16* __restrict__ ctxb, const __bf16* __restrict__ wob,
    const float* __restrict__ bo, float* __restrict__ out) {
  const int lane = threadIdx.x & 31;
  const int wave = threadIdx.x >> 5;
  const int m0 = (blockIdx.x * 4 + wave) * 16;
  const int n0 = blockIdx.y * 64;

  v8f acc[4];
#pragma unroll
  for (int s = 0; s < 4; ++s) acc[s] = vzero8();

#pragma unroll 2
  for (int k0 = 0; k0 < D_MODEL; k0 += 32) {
    v16bf a = load_a_frag(ctxb, D_MODEL, m0, k0, lane);
#pragma unroll
    for (int s = 0; s < 4; ++s) {
      v16bf b = load_b_frag(wob, D_MODEL, n0 + s * 16, k0, lane);
      acc[s] = wmma_bf16(a, b, acc[s]);
    }
  }
  const int c = lane & 15, g = lane >> 4;
#pragma unroll
  for (int s = 0; s < 4; ++s) {
    const float bn = bo[n0 + s * 16 + c];
#pragma unroll
    for (int vr = 0; vr < 8; ++vr)
      out[(size_t)(m0 + vr + 8 * g) * D_MODEL + n0 + s * 16 + c] =
          acc[s][vr] + bn;
  }
}

// ---------------------------------------------------------------------------
extern "C" void kernel_launch(void* const* d_in, const int* in_sizes, int n_in,
                              void* d_out, int out_size, void* d_ws, size_t ws_size,
                              hipStream_t stream) {
  (void)in_sizes; (void)n_in; (void)out_size; (void)ws_size;
  const float* x  = (const float*)d_in[0];
  const float* Wq = (const float*)d_in[1];
  const float* bq = (const float*)d_in[2];
  const float* Wk = (const float*)d_in[3];
  const float* bk = (const float*)d_in[4];
  const float* Wv = (const float*)d_in[5];
  const float* bv = (const float*)d_in[6];
  const float* Wo = (const float*)d_in[7];
  const float* bo = (const float*)d_in[8];
  float* out = (float*)d_out;

  const size_t XE = (size_t)MROWS * D_MODEL;
  const size_t WE = (size_t)D_MODEL * D_MODEL;
  char* ws = (char*)d_ws;
  __bf16* xb   = (__bf16*)ws; ws += XE * sizeof(__bf16);
  __bf16* wqb  = (__bf16*)ws; ws += WE * sizeof(__bf16);
  __bf16* wkb  = (__bf16*)ws; ws += WE * sizeof(__bf16);
  __bf16* wvb  = (__bf16*)ws; ws += WE * sizeof(__bf16);
  __bf16* wob  = (__bf16*)ws; ws += WE * sizeof(__bf16);
  __bf16* Qb   = (__bf16*)ws; ws += XE * sizeof(__bf16);
  __bf16* Kb   = (__bf16*)ws; ws += XE * sizeof(__bf16);
  __bf16* Vtb  = (__bf16*)ws; ws += XE * sizeof(__bf16);
  __bf16* ctxb = (__bf16*)ws; ws += XE * sizeof(__bf16);

  cvt_kernel<<<(int)(XE / 4 / 256), 256, 0, stream>>>(x,  xb,  (int)(XE / 4));
  cvt_kernel<<<(int)(WE / 4 / 256), 256, 0, stream>>>(Wq, wqb, (int)(WE / 4));
  cvt_kernel<<<(int)(WE / 4 / 256), 256, 0, stream>>>(Wk, wkb, (int)(WE / 4));
  cvt_kernel<<<(int)(WE / 4 / 256), 256, 0, stream>>>(Wv, wvb, (int)(WE / 4));
  cvt_kernel<<<(int)(WE / 4 / 256), 256, 0, stream>>>(Wo, wob, (int)(WE / 4));

  qkv_kernel<<<dim3(MROWS / 64, D_MODEL / 64, 3), 128, 0, stream>>>(
      xb, wqb, wkb, wvb, bq, bk, bv, Qb, Kb, Vtb);

  attn_kernel<<<dim3(S_LEN / 128, BATCH * NHEAD), 256, 0, stream>>>(
      Qb, Kb, Vtb, ctxb);

  oproj_kernel<<<dim3(MROWS / 64, D_MODEL / 64), 128, 0, stream>>>(
      ctxb, wob, bo, out);
}
